// SelectiveSSM_39419209843085
// MI455X (gfx1250) — matrix operand order repaired and checked by reading, not compiled, not run on hardware
//
#include <hip/hip_runtime.h>
#include <hip/hip_bf16.h>
#include <math.h>

// ---------------- problem constants ----------------
#define BB      2
#define TT      2048
#define DM      1024
#define NS      16
#define ROWS    (BB * TT)          // 4096
#define E_PROJ  (2 * NS + 1)       // 33

// workspace layout (in floats)
#define WS_XZ    0u                         // [ROWS][2*DM]
#define WS_XCONV (WS_XZ + ROWS * 2 * DM)    // [ROWS][DM]
#define WS_SSM   (WS_XCONV + ROWS * DM)     // [ROWS][33]
#define WS_Y     (WS_SSM + ROWS * E_PROJ)   // [ROWS][DM]

typedef __attribute__((ext_vector_type(16))) __bf16 v16bf;
typedef __attribute__((ext_vector_type(8)))  __bf16 v8bf;
typedef __attribute__((ext_vector_type(8)))  float  v8f;
typedef __attribute__((ext_vector_type(4)))  float  f4;
typedef __attribute__((ext_vector_type(4)))  unsigned int u32x4;
typedef __attribute__((ext_vector_type(4)))  int    i32x4;
typedef __attribute__((ext_vector_type(8)))  int    i32x8;

// ---------------- LDS layout for the GEMM (bytes) ----------------
// fp32 staging tiles: 128 rows x 32 floats, TDM-padded to stride 36 DW (=144B)
#define F32_TILE_B   (128 * 36 * 4)            // 18432
// bf16 compute tiles: 128 rows x 32 bf16, stride 40 elems (=80B, 16B aligned)
#define BF16_TILE_B  (128 * 40 * 2)            // 10240
#define LDS_F32A(buf) ((buf) * 2 * F32_TILE_B)
#define LDS_F32B(buf) ((buf) * 2 * F32_TILE_B + F32_TILE_B)
#define LDS_BF16A     (4 * F32_TILE_B)
#define LDS_BF16B     (4 * F32_TILE_B + BF16_TILE_B)
#define LDS_TOTAL_B   (4 * F32_TILE_B + 2 * BF16_TILE_B)   // 94208

// ============================================================================
// Tensor Data Mover: load one [128 rows x 32 floats] 2D tile into LDS, with
// LDS padding of 4 DWORDs after every 32 DWORDs (row stride 36 DW).
// D# fields per CDNA5 ISA ch.8: group0 = {count,lds_addr,global_addr,type},
// group1 = {data_size,pad cfg, tensor dims, tile dims, dim0 stride}.
// ============================================================================
__device__ __forceinline__ void tdm_load_tile(const float* gsrc, int K, int nrows,
                                              unsigned lds_byte_addr)
{
    unsigned long long ga = (unsigned long long)(uintptr_t)gsrc;
    u32x4 g0;
    g0[0] = 1u;                                            // count = 1 (valid D#)
    g0[1] = lds_byte_addr;                                 // LDS dest (bytes)
    g0[2] = (unsigned)(ga & 0xffffffffu);                  // global_addr[31:0]
    g0[3] = (unsigned)((ga >> 32) & 0x01ffffffu)           // global_addr[56:32]
          | (2u << 30);                                    // type = 2 ("image")
    i32x8 g1;
    g1[0] = (int)((2u << 16)        // data_size = 4 bytes
          |       (1u << 20)        // pad_enable
          |       (4u << 22)        // pad_interval: code 4 -> every 32 DWORDs
          |       (3u << 25));      // pad_amount:   code 3 -> 4 DWORDs
    g1[1] = (int)(((unsigned)K & 0xffffu) << 16);          // tensor_dim0[15:0]
    g1[2] = (int)(((unsigned)K >> 16)
          |      (((unsigned)nrows & 0xffffu) << 16));     // dim0 hi | dim1 lo
    g1[3] = (int)((((unsigned)nrows >> 16) & 0xffffu)
          |       (32u << 16));                            // dim1 hi | tile_dim0=32
    g1[4] = 128;                                           // tile_dim1=128, tile_dim2=0
    g1[5] = K;                                             // tensor_dim0_stride lo32
    g1[6] = 0;
    g1[7] = 0;
    i32x4 z4 = {0, 0, 0, 0};
#if __clang_major__ >= 23
    i32x8 z8 = {0, 0, 0, 0, 0, 0, 0, 0};
    __builtin_amdgcn_tensor_load_to_lds(g0, g1, z4, z4, z8, 0);
#else
    __builtin_amdgcn_tensor_load_to_lds(g0, g1, z4, z4, 0);
#endif
}

// ============================================================================
// WMMA GEMM:  C[M,N] = A[M,K] @ W[N,K]^T  (fp32 in/out, bf16 WMMA compute)
// Block = 256 threads = 8 waves -> 128x128 C tile. K stepped by 32.
// Pipeline per step: TDM prefetch (s+1) | convert fp32->bf16 tiles | 8 WMMAs.
// Fragment layouts per ISA 7.12.2 (h = lane>>4, m/n = lane&15):
//   A elems 0..7  <- A[m][k0+8h..], elems 8..15 <- A[m][k0+16+8h..]
//   B elems 0..7  <- W[n][k0+8h..], elems 8..15 <- W[n][k0+16+8h..]
//   C vgpr r -> C[r+8h][n]
// ============================================================================
__global__ __launch_bounds__(256) void gemm_tdm_wmma_kernel(
    const float* __restrict__ A, const float* __restrict__ W,
    float* __restrict__ C, int M, int N, int K, int nblk_n)
{
    extern __shared__ char smem[];
    const int tid  = threadIdx.x;
    const int wave = tid >> 5;
    const int lane = tid & 31;
    const int mb   = blockIdx.x / nblk_n;
    const int nb   = blockIdx.x - mb * nblk_n;
    const int wr   = wave >> 2;            // 0..1 : 64-row strip
    const int wc   = wave & 3;             // 0..3 : 32-col strip
    const int mloc = lane & 15;
    const int half = lane >> 4;

    const unsigned ldsbase = (unsigned)(uintptr_t)&smem[0];
    const float* Abase = A + (size_t)mb * 128 * K;
    const float* Wbase = W + (size_t)nb * 128 * K;
    const int steps = K >> 5;

    v8f acc[4][2] = {};

    // prologue: stage tiles for k-step 0 into fp32 buffer 0
    if (tid < 32) {
        tdm_load_tile(Abase, K, M, ldsbase + LDS_F32A(0));
        tdm_load_tile(Wbase, K, N, ldsbase + LDS_F32B(0));
    }

    for (int s = 0; s < steps; ++s) {
        const int cur = s & 1;
        if (tid < 32) {
            if (s + 1 < steps) {           // async prefetch of next K tiles
                const float* an = Abase + ((s + 1) << 5);
                const float* wn = Wbase + ((s + 1) << 5);
                tdm_load_tile(an, K, M, ldsbase + LDS_F32A(cur ^ 1));
                tdm_load_tile(wn, K, N, ldsbase + LDS_F32B(cur ^ 1));
                __builtin_amdgcn_s_wait_tensorcnt(2);   // oldest 2 (step s) done
            } else {
                __builtin_amdgcn_s_wait_tensorcnt(0);
            }
        }
        __syncthreads();                   // fp32 tiles for step s visible

        // cooperative fp32 -> bf16 tile conversion (one cvt per element)
        {
            const float*  fA = (const float*)(smem + LDS_F32A(cur));
            const float*  fB = (const float*)(smem + LDS_F32B(cur));
            __bf16* hA = (__bf16*)(smem + LDS_BF16A);
            __bf16* hB = (__bf16*)(smem + LDS_BF16B);
            const int row = tid >> 1;              // 0..127
            const int hr  = (tid & 1) << 4;        // 0 / 16 floats
            const float* pa = fA + row * 36 + hr;
            const float* pb = fB + row * 36 + hr;
            f4 a0 = *(const f4*)(pa);     f4 a1 = *(const f4*)(pa + 4);
            f4 a2 = *(const f4*)(pa + 8); f4 a3 = *(const f4*)(pa + 12);
            f4 b0 = *(const f4*)(pb);     f4 b1 = *(const f4*)(pb + 4);
            f4 b2 = *(const f4*)(pb + 8); f4 b3 = *(const f4*)(pb + 12);
            v8bf alo, ahi, blo, bhi;
#pragma unroll
            for (int i = 0; i < 4; ++i) {
                alo[i] = (__bf16)a0[i]; alo[4 + i] = (__bf16)a1[i];
                ahi[i] = (__bf16)a2[i]; ahi[4 + i] = (__bf16)a3[i];
                blo[i] = (__bf16)b0[i]; blo[4 + i] = (__bf16)b1[i];
                bhi[i] = (__bf16)b2[i]; bhi[4 + i] = (__bf16)b3[i];
            }
            *(v8bf*)(hA + row * 40 + hr)     = alo;
            *(v8bf*)(hA + row * 40 + hr + 8) = ahi;
            *(v8bf*)(hB + row * 40 + hr)     = blo;
            *(v8bf*)(hB + row * 40 + hr + 8) = bhi;
        }
        __syncthreads();                   // bf16 tiles ready

        // fragment loads (pure ds_load_b128) + 8 WMMAs
        const __bf16* tA = (const __bf16*)(smem + LDS_BF16A);
        const __bf16* tB = (const __bf16*)(smem + LDS_BF16B);
        v16bf afrag[4], bfrag[2];
#pragma unroll
        for (int i = 0; i < 4; ++i) {
            const __bf16* p = tA + (wr * 64 + i * 16 + mloc) * 40 + half * 8;
            v8bf r0 = *(const v8bf*)(p);
            v8bf r1 = *(const v8bf*)(p + 16);
            afrag[i] = __builtin_shufflevector(r0, r1, 0, 1, 2, 3, 4, 5, 6, 7,
                                               8, 9, 10, 11, 12, 13, 14, 15);
        }
#pragma unroll
        for (int j = 0; j < 2; ++j) {
            const __bf16* p = tB + (wc * 32 + j * 16 + mloc) * 40 + half * 8;
            v8bf r0 = *(const v8bf*)(p);
            v8bf r1 = *(const v8bf*)(p + 16);
            bfrag[j] = __builtin_shufflevector(r0, r1, 0, 1, 2, 3, 4, 5, 6, 7,
                                               8, 9, 10, 11, 12, 13, 14, 15);
        }
#pragma unroll
        for (int i = 0; i < 4; ++i)
#pragma unroll
            for (int j = 0; j < 2; ++j)
                acc[i][j] = __builtin_amdgcn_wmma_f32_16x16x32_bf16(
                    false, afrag[i], false, bfrag[j],
                    (short)0, acc[i][j], false, false);
        __syncthreads();                   // done with fp32 cur before re-issue
    }

    // store 128x128 C tile
#pragma unroll
    for (int i = 0; i < 4; ++i)
#pragma unroll
        for (int j = 0; j < 2; ++j) {
            float* crow = C + (size_t)(mb * 128 + wr * 64 + i * 16 + half * 8) * N
                            + nb * 128 + wc * 32 + j * 16 + mloc;
#pragma unroll
            for (int r = 0; r < 8; ++r)
                crow[(size_t)r * N] = acc[i][j][r];
        }
}

// ============================================================================
// Depthwise causal conv (4 taps) + bias + SiLU.
// ============================================================================
__global__ __launch_bounds__(256) void conv_silu_kernel(
    const float* __restrict__ xz, const float* __restrict__ cw,
    const float* __restrict__ cb, float* __restrict__ xc)
{
    int idx = blockIdx.x * 256 + threadIdx.x;        // over ROWS*DM
    int row = idx >> 10;                             // b*TT + t
    int d   = idx & (DM - 1);
    int t   = row & (TT - 1);

    const float* base = xz + (size_t)row * (2 * DM) + d;
    const float* w = cw + d * 4;

    float acc = cb[d] + w[3] * base[0];
    if (t >= 1) acc += w[2] * base[-(ptrdiff_t)(2 * DM)];
    if (t >= 2) acc += w[1] * base[-(ptrdiff_t)(4 * DM)];
    if (t >= 3) acc += w[0] * base[-(ptrdiff_t)(6 * DM)];

    float sig = 1.0f / (1.0f + __expf(-acc));        // SiLU
    xc[(size_t)row * DM + d] = acc * sig;
}

// ============================================================================
// x_proj: ssm[row][e] = dot(x_conv[row][:], x_proj_w[e][:]), e in [0,33).
// One wave32 per (row, e).
// ============================================================================
__global__ __launch_bounds__(256) void xproj_kernel(
    const float* __restrict__ xc, const float* __restrict__ pw,
    float* __restrict__ sp)
{
    int gwave = (blockIdx.x * 256 + threadIdx.x) >> 5;
    int lane  = threadIdx.x & 31;
    int row   = gwave / E_PROJ;
    int e     = gwave - row * E_PROJ;

    const float* xr = xc + (size_t)row * DM;
    const float* wr = pw + (size_t)e * DM;

    float s = 0.0f;
#pragma unroll 8
    for (int j = lane; j < DM; j += 32)
        s += xr[j] * wr[j];
#pragma unroll
    for (int o = 16; o > 0; o >>= 1)
        s += __shfl_xor(s, o, 32);
    if (lane == 0)
        sp[(size_t)row * E_PROJ + e] = s;
}

// ============================================================================
// Selective scan: one thread per (b,d); h[16] in VGPRs.
// ============================================================================
__global__ __launch_bounds__(256) void scan_kernel(
    const float* __restrict__ xz, const float* __restrict__ xc,
    const float* __restrict__ sp, const float* __restrict__ dtw,
    const float* __restrict__ dtb, const float* __restrict__ Alog,
    float* __restrict__ y)
{
    int id = blockIdx.x * 256 + threadIdx.x;         // 0 .. BB*DM-1
    int b = id >> 10;
    int d = id & (DM - 1);

    float Ar[NS];
#pragma unroll
    for (int n = 0; n < NS; ++n)
        Ar[n] = -__expf(Alog[d * NS + n]);

    const float dw = dtw[d];
    const float db = dtb[d];

    float h[NS];
#pragma unroll
    for (int n = 0; n < NS; ++n) h[n] = 0.0f;

    for (int t = 0; t < TT; ++t) {
        size_t row = (size_t)b * TT + t;
        const float* s = sp + row * E_PROJ;

        float pre = s[E_PROJ - 1] * dw + db;
        float dt  = (pre > 20.0f) ? pre : log1pf(__expf(pre));   // softplus

        float xcv = xc[row * DM + d];
        float zv  = xz[row * (2 * DM) + DM + d];
        float dtx = dt * xcv;

        float acc = 0.0f;
#pragma unroll
        for (int n = 0; n < NS; ++n) {
            float dA = __expf(dt * Ar[n]);
            h[n] = dA * h[n] + dtx * s[n];
            acc += h[n] * s[NS + n];
        }

        float sig = 1.0f / (1.0f + __expf(-zv));     // silu(z)
        y[row * DM + d] = acc * (zv * sig);
    }
}

// ============================================================================
extern "C" void kernel_launch(void* const* d_in, const int* in_sizes, int n_in,
                              void* d_out, int out_size, void* d_ws, size_t ws_size,
                              hipStream_t stream) {
    const float* x      = (const float*)d_in[0];   // [B,T,DM]
    const float* ipw    = (const float*)d_in[1];   // [2*DM, DM]
    const float* convw  = (const float*)d_in[2];   // [DM,1,4]
    const float* convb  = (const float*)d_in[3];   // [DM]
    const float* xpw    = (const float*)d_in[4];   // [33, DM]
    const float* dtw    = (const float*)d_in[5];   // [DM,1]
    const float* dtb    = (const float*)d_in[6];   // [DM]
    const float* Alog   = (const float*)d_in[7];   // [DM,NS]
    const float* opw    = (const float*)d_in[8];   // [DM, DM]
    float*       out    = (float*)d_out;           // [B,T,DM]

    float* ws     = (float*)d_ws;
    float* ws_xz  = ws + WS_XZ;
    float* ws_xc  = ws + WS_XCONV;
    float* ws_sp  = ws + WS_SSM;
    float* ws_y   = ws + WS_Y;

    // 1) in_proj: xz[4096,2048] = x @ in_proj_w^T   (32 x 16 blocks of 128x128)
    {
        int nblk_n = (2 * DM) / 128;
        int blocks = (ROWS / 128) * nblk_n;
        gemm_tdm_wmma_kernel<<<blocks, 256, LDS_TOTAL_B, stream>>>(
            x, ipw, ws_xz, ROWS, 2 * DM, DM, nblk_n);
    }
    // 2) causal depthwise conv + SiLU
    conv_silu_kernel<<<(ROWS * DM) / 256, 256, 0, stream>>>(ws_xz, convw, convb, ws_xc);
    // 3) x_proj (33 outputs per row)
    xproj_kernel<<<(ROWS * E_PROJ) / 8, 256, 0, stream>>>(ws_xc, xpw, ws_sp);
    // 4) selective scan + gating
    scan_kernel<<<(BB * DM) / 256, 256, 0, stream>>>(ws_xz, ws_xc, ws_sp,
                                                     dtw, dtb, Alog, ws_y);
    // 5) out_proj: out[4096,1024] = y @ out_proj_w^T  (32 x 8 blocks of 128x128)
    {
        int nblk_n = DM / 128;
        int blocks = (ROWS / 128) * nblk_n;
        gemm_tdm_wmma_kernel<<<blocks, 256, LDS_TOTAL_B, stream>>>(
            ws_y, opw, out, ROWS, DM, DM, nblk_n);
    }
}
